// fuT_35588099015111
// MI455X (gfx1250) — matrix operand order repaired and checked by reading, not compiled, not run on hardware
//
#include <hip/hip_runtime.h>
#include <hip/hip_bf16.h>

typedef __attribute__((ext_vector_type(16))) __bf16 v16bf;
typedef __attribute__((ext_vector_type(8)))  float  v8f;

#define B_    16
#define N_    1024
#define DIN_  272
#define HID_  256
#define KV_   4
#define DFF_  1024

#define LDSP  40   // padded LDS row stride in bf16 (80B: 16B-aligned, bank-spreading)

union FragU { uint4 u4[2]; v16bf v; };

// ---------------------------------------------------------------------------
// Tiled GEMM: C[M x N] = A[M x K] * Bm[K x N] + bias (+ReLU)
// A accessed via generalized strides (handles NCHW token reads without a
// transpose pass):  a(m,k) = A[(m/aMb)*aBatchStride + (m%aMb)*aRowStride + k*aColStride]
// 256 threads / 8 waves; block tile 128(M) x 64(N); waves 4(M) x 2(N); each
// wave computes 2x2 16x16 WMMA tiles (4 WMMAs per K-step from 2 A + 2 B frags).
// Branch-free main K loop (tail peeled; OOB handled by address clamp + select,
// never by EXEC-masked loads). Fragments are aligned 16B LDS reads matching
// the CDNA5 ISA 7.12.2 bf16 lane layouts.
// ---------------------------------------------------------------------------
__global__ __launch_bounds__(256) void gemm_wmma_bf16(
    const float* __restrict__ A, int aMb, long long aBatchStride,
    long long aRowStride, long long aColStride,
    const float* __restrict__ Bm, int ldb,
    const float* __restrict__ bias,
    float* __restrict__ C, int ldc,
    int M, int N, int Kdim, int relu)
{
  __shared__ __bf16 As[128 * LDSP];  // [row][k]
  __shared__ __bf16 BsT[64 * LDSP];  // [n][k] (transposed)

  const int t    = threadIdx.x;
  const int lane = t & 31;
  const int wid  = t >> 5;
  const int half = lane >> 4;
  const int l16  = lane & 15;

  const int waveM = wid & 3;   // 0..3 -> M 32-row strip
  const int waveN = wid >> 2;  // 0..1 -> N 32-col strip

  const int bm0 = blockIdx.y * 128;
  const int bn0 = blockIdx.x * 64;

  // ---- staging assignments ----
  // A: thread t -> row (t>>1), 16 consecutive k at (t&1)*16
  const int arow = t >> 1;
  const int akb  = (t & 1) * 16;
  const int garow = bm0 + arow;
  const float* aptr = A +
      (long long)(garow / aMb) * aBatchStride +
      (long long)(garow % aMb) * aRowStride;
  // B: thread t -> k row (t>>3), 8 consecutive n at (t&7)*8
  const int bkr = t >> 3;
  const int bnb = (t & 7) * 8;
  // clamp the 8-wide column run so reads stay in-bounds for ragged N;
  // duplicated values land in columns that are never stored.
  const int bncl = (bn0 + bnb + 8 <= N) ? (bn0 + bnb) : (N - 8);

  v8f acc00 = {}, acc01 = {}, acc10 = {}, acc11 = {};

  const int Kfull = Kdim & ~31;

  // ================= branch-free main K loop =================
  for (int k0 = 0; k0 < Kfull; k0 += 32) {
    {  // stage A (128 x 32)
      union { __bf16 h[16]; uint4 u4[4]; } pk;
      const float* ap = aptr + (long long)(k0 + akb) * aColStride;
#pragma unroll
      for (int i = 0; i < 16; ++i)
        pk.h[i] = (__bf16)ap[(long long)i * aColStride];
      uint4* dst = (uint4*)&As[arow * LDSP + akb];
#pragma unroll
      for (int i = 0; i < 4; ++i) dst[i] = pk.u4[i];
    }
    {  // stage B (32 x 64) transposed
      const float* bp = Bm + (long long)(k0 + bkr) * ldb + bncl;
#pragma unroll
      for (int i = 0; i < 8; ++i)
        BsT[(bnb + i) * LDSP + bkr] = (__bf16)bp[i];
    }
    __syncthreads();
    {
      FragU fa0, fa1, fb0, fb1;
      const int ar = waveM * 32 + l16;
      fa0.u4[0] = *(const uint4*)&As[ar * LDSP + 8 * half];
      fa0.u4[1] = *(const uint4*)&As[ar * LDSP + 16 + 8 * half];
      fa1.u4[0] = *(const uint4*)&As[(ar + 16) * LDSP + 8 * half];
      fa1.u4[1] = *(const uint4*)&As[(ar + 16) * LDSP + 16 + 8 * half];
      const int bn = waveN * 32 + l16;
      fb0.u4[0] = *(const uint4*)&BsT[bn * LDSP + 8 * half];
      fb0.u4[1] = *(const uint4*)&BsT[bn * LDSP + 16 + 8 * half];
      fb1.u4[0] = *(const uint4*)&BsT[(bn + 16) * LDSP + 8 * half];
      fb1.u4[1] = *(const uint4*)&BsT[(bn + 16) * LDSP + 16 + 8 * half];

      acc00 = __builtin_amdgcn_wmma_f32_16x16x32_bf16(false, fa0.v, false, fb0.v, (short)0, acc00, false, false);
      acc01 = __builtin_amdgcn_wmma_f32_16x16x32_bf16(false, fa0.v, false, fb1.v, (short)0, acc01, false, false);
      acc10 = __builtin_amdgcn_wmma_f32_16x16x32_bf16(false, fa1.v, false, fb0.v, (short)0, acc10, false, false);
      acc11 = __builtin_amdgcn_wmma_f32_16x16x32_bf16(false, fa1.v, false, fb1.v, (short)0, acc11, false, false);
    }
    __syncthreads();
  }

  // ================= peeled K tail (executes at most once) =================
  if (Kfull < Kdim) {
    const int k0 = Kfull;
    {  // stage A with clamp + select-zero (no EXEC-masked loads)
      union { __bf16 h[16]; uint4 u4[4]; } pk;
#pragma unroll
      for (int i = 0; i < 16; ++i) {
        int k = k0 + akb + i;
        int kc = (k < Kdim) ? k : (Kdim - 1);
        float v = aptr[(long long)kc * aColStride];
        pk.h[i] = (__bf16)((k < Kdim) ? v : 0.0f);
      }
      uint4* dst = (uint4*)&As[arow * LDSP + akb];
#pragma unroll
      for (int i = 0; i < 4; ++i) dst[i] = pk.u4[i];
    }
    {  // stage B with k clamp (A-side zeros nullify the garbage products)
      const int kg = k0 + bkr;
      const int kgc = (kg < Kdim) ? kg : (Kdim - 1);
      const float* bp = Bm + (long long)kgc * ldb + bncl;
#pragma unroll
      for (int i = 0; i < 8; ++i)
        BsT[(bnb + i) * LDSP + bkr] = (__bf16)bp[i];
    }
    __syncthreads();
    {
      FragU fa0, fa1, fb0, fb1;
      const int ar = waveM * 32 + l16;
      fa0.u4[0] = *(const uint4*)&As[ar * LDSP + 8 * half];
      fa0.u4[1] = *(const uint4*)&As[ar * LDSP + 16 + 8 * half];
      fa1.u4[0] = *(const uint4*)&As[(ar + 16) * LDSP + 8 * half];
      fa1.u4[1] = *(const uint4*)&As[(ar + 16) * LDSP + 16 + 8 * half];
      const int bn = waveN * 32 + l16;
      fb0.u4[0] = *(const uint4*)&BsT[bn * LDSP + 8 * half];
      fb0.u4[1] = *(const uint4*)&BsT[bn * LDSP + 16 + 8 * half];
      fb1.u4[0] = *(const uint4*)&BsT[(bn + 16) * LDSP + 8 * half];
      fb1.u4[1] = *(const uint4*)&BsT[(bn + 16) * LDSP + 16 + 8 * half];

      acc00 = __builtin_amdgcn_wmma_f32_16x16x32_bf16(false, fa0.v, false, fb0.v, (short)0, acc00, false, false);
      acc01 = __builtin_amdgcn_wmma_f32_16x16x32_bf16(false, fa0.v, false, fb1.v, (short)0, acc01, false, false);
      acc10 = __builtin_amdgcn_wmma_f32_16x16x32_bf16(false, fa1.v, false, fb0.v, (short)0, acc10, false, false);
      acc11 = __builtin_amdgcn_wmma_f32_16x16x32_bf16(false, fa1.v, false, fb1.v, (short)0, acc11, false, false);
    }
  }

  // ---- epilogue: bias (+ReLU) and store ----
#pragma unroll
  for (int mi = 0; mi < 2; ++mi) {
#pragma unroll
    for (int ni = 0; ni < 2; ++ni) {
      const int col = bn0 + waveN * 32 + ni * 16 + l16;
      if (col < N) {
        const float bcol = bias[col];
        const v8f accv = mi ? (ni ? acc11 : acc10) : (ni ? acc01 : acc00);
#pragma unroll
        for (int v = 0; v < 8; ++v) {
          int row = bm0 + waveM * 32 + mi * 16 + v + 8 * half;
          float r = accv[v] + bcol;
          if (relu) r = fmaxf(r, 0.0f);
          C[(long long)row * ldc + col] = r;
        }
      }
    }
  }
}

// ---------------------------------------------------------------------------
// Positional embedding: pe[(i*32+j)*256 + c] = c<128 ? col[j][c] : row[i][c-128]
// ---------------------------------------------------------------------------
__global__ __launch_bounds__(256) void pe_kernel(
    const float* __restrict__ re, const float* __restrict__ ce,
    float* __restrict__ pe)
{
  int idx = blockIdx.x * 256 + threadIdx.x;   // 1024*256 total
  int n = idx >> 8, c = idx & 255;
  int i = n >> 5, j = n & 31;
  pe[idx] = (c < 128) ? ce[j * 128 + c] : re[i * 128 + (c - 128)];
}

// ---------------------------------------------------------------------------
// Local-window (7x7) attention, exploiting the -inf neighbor mask.
// One 256-thread block per (b, n). o[b][n][k*256+c].
// ---------------------------------------------------------------------------
__global__ __launch_bounds__(256) void attn_kernel(
    const float* __restrict__ x, const float* __restrict__ kt,
    const float* __restrict__ pe, float* __restrict__ o)
{
  const int n = blockIdx.x;
  const int b = blockIdx.y;
  const int t = threadIdx.x;
  const int lane = t & 31, wid = t >> 5;
  const int r = n >> 5, c = n & 31;
  const int r0 = (r - 3 > 0) ? r - 3 : 0;
  const int r1 = (r + 3 < 31) ? r + 3 : 31;
  const int c0 = (c - 3 > 0) ? c - 3 : 0;
  const int c1 = (c + 3 < 31) ? c + 3 : 31;
  const int Wn = c1 - c0 + 1;
  const int nh = (r1 - r0 + 1) * Wn;   // <= 49

  __shared__ float qp[HID_];
  __shared__ float sc[KV_][64];
  const float scale = 0.0625f;   // 1/sqrt(256)

  qp[t] = (x[((long long)(b * N_ + n)) * HID_ + t] + pe[n * HID_ + t]) * scale;
  __syncthreads();

  // scores: wave per (k, neighbor) job
  for (int j = wid; j < KV_ * nh; j += 8) {
    int k = j / nh, idx = j % nh;
    int m = (r0 + idx / Wn) * 32 + (c0 + idx % Wn);
    const float* kr = kt + ((long long)((k * B_ + b) * N_ + m)) * HID_;
    const float* pr = pe + m * HID_;
    float s = 0.0f;
    for (int cc = lane; cc < HID_; cc += 32)
      s += qp[cc] * (kr[cc] + pr[cc]);
    for (int off = 16; off > 0; off >>= 1) s += __shfl_xor(s, off, 32);
    if (lane == 0) sc[k][idx] = s;
  }
  __syncthreads();

  // softmax per k (waves 0..3)
  if (wid < KV_) {
    float mx = -3.4e38f;
    for (int i = lane; i < nh; i += 32) mx = fmaxf(mx, sc[wid][i]);
    for (int off = 16; off > 0; off >>= 1) mx = fmaxf(mx, __shfl_xor(mx, off, 32));
    float sum = 0.0f;
    for (int i = lane; i < nh; i += 32) {
      float e = __expf(sc[wid][i] - mx);
      sc[wid][i] = e;
      sum += e;
    }
    for (int off = 16; off > 0; off >>= 1) sum += __shfl_xor(sum, off, 32);
    float inv = 1.0f / sum;
    for (int i = lane; i < nh; i += 32) sc[wid][i] *= inv;
  }
  __syncthreads();

  // o = attn @ kt (values are the raw matched tokens)
  float* orow = o + ((long long)(b * N_ + n)) * (KV_ * HID_);
  for (int k = 0; k < KV_; ++k) {
    const float* kb = kt + ((long long)((k * B_ + b) * N_)) * HID_;
    float accv = 0.0f;
    for (int idx = 0; idx < nh; ++idx) {
      int m = (r0 + idx / Wn) * 32 + (c0 + idx % Wn);
      accv += sc[k][idx] * kb[(long long)m * HID_ + t];
    }
    orow[k * HID_ + t] = accv;
  }
}

// ---------------------------------------------------------------------------
// x = LayerNorm(x + add) * g + b  — one block per token row (HID=256)
// ---------------------------------------------------------------------------
__global__ __launch_bounds__(256) void add_ln_kernel(
    float* __restrict__ x, const float* __restrict__ add,
    const float* __restrict__ g, const float* __restrict__ bb)
{
  long long row = blockIdx.x;
  int t = threadIdx.x;
  float v = x[row * HID_ + t] + add[row * HID_ + t];
  __shared__ float s1[256], s2[256];
  s1[t] = v; s2[t] = v * v;
  __syncthreads();
  for (int off = 128; off > 0; off >>= 1) {
    if (t < off) { s1[t] += s1[t + off]; s2[t] += s2[t + off]; }
    __syncthreads();
  }
  float mean = s1[0] * (1.0f / 256.0f);
  float var  = s2[0] * (1.0f / 256.0f) - mean * mean;
  float inv  = rsqrtf(var + 1e-5f);
  x[row * HID_ + t] = (v - mean) * inv * g[t] + bb[t];
}

// ---------------------------------------------------------------------------
// recon transpose to NCHW + per-pixel L2 anomaly map
// ---------------------------------------------------------------------------
__global__ __launch_bounds__(256) void recon_kernel(
    const float* __restrict__ tok, const float* __restrict__ qf,
    float* __restrict__ recon_out, float* __restrict__ pred0)
{
  int hw = blockIdx.x, b = blockIdx.y, t = threadIdx.x;
  __shared__ float s[256];
  float acc = 0.0f;
  long long tbase = ((long long)(b * N_ + hw)) * DIN_;
  long long qbase = (long long)b * DIN_ * N_ + hw;
  for (int cc = t; cc < DIN_; cc += 256) {
    float rv = tok[tbase + cc];
    float qv = qf[qbase + (long long)cc * N_];
    recon_out[qbase + (long long)cc * N_] = rv;
    float d = rv - qv;
    acc += d * d;
  }
  s[t] = acc;
  __syncthreads();
  for (int off = 128; off > 0; off >>= 1) {
    if (t < off) s[t] += s[t + off];
    __syncthreads();
  }
  if (t == 0) pred0[b * N_ + hw] = sqrtf(s[0]);
}

__global__ __launch_bounds__(256) void minmax_kernel(
    const float* __restrict__ p, float* __restrict__ mm)
{
  int t = threadIdx.x;
  float mn = 3.4e38f, mx = -3.4e38f;
  for (int i = t; i < B_ * N_; i += 256) {
    float v = p[i];
    mn = fminf(mn, v);
    mx = fmaxf(mx, v);
  }
  __shared__ float smn[256], smx[256];
  smn[t] = mn; smx[t] = mx;
  __syncthreads();
  for (int off = 128; off > 0; off >>= 1) {
    if (t < off) {
      smn[t] = fminf(smn[t], smn[t + off]);
      smx[t] = fmaxf(smx[t], smx[t + off]);
    }
    __syncthreads();
  }
  if (t == 0) { mm[0] = smn[0]; mm[1] = smx[0]; }
}

// normalize + align_corners bilinear 32x32 -> 256x256
__global__ __launch_bounds__(256) void resize_kernel(
    const float* __restrict__ pred0, const float* __restrict__ mm,
    float* __restrict__ out)
{
  int idx = blockIdx.x * 256 + threadIdx.x;   // 16*256*256
  int b = idx >> 16;
  int p = idx & 65535;
  int y = p >> 8, xq = p & 255;
  float mn = mm[0];
  float inv = 1.0f / (mm[1] - mn);
  const float sc = 31.0f / 255.0f;
  float ys = y * sc, xs = xq * sc;
  int y0 = (int)ys, x0 = (int)xs;
  int y1 = (y0 + 1 < 31) ? y0 + 1 : 31;
  int x1 = (x0 + 1 < 31) ? x0 + 1 : 31;
  float fy = ys - y0, fx = xs - x0;
  const float* pb = pred0 + b * N_;
  float v00 = (pb[y0 * 32 + x0] - mn) * inv;
  float v01 = (pb[y0 * 32 + x1] - mn) * inv;
  float v10 = (pb[y1 * 32 + x0] - mn) * inv;
  float v11 = (pb[y1 * 32 + x1] - mn) * inv;
  out[idx] = (v00 * (1.0f - fx) + v01 * fx) * (1.0f - fy) +
             (v10 * (1.0f - fx) + v11 * fx) * fy;
}

// ---------------------------------------------------------------------------
extern "C" void kernel_launch(void* const* d_in, const int* in_sizes, int n_in,
                              void* d_out, int out_size, void* d_ws, size_t ws_size,
                              hipStream_t stream)
{
  const float* qf   = (const float*)d_in[0];
  const float* mf   = (const float*)d_in[1];
  const float* qW   = (const float*)d_in[2];
  const float* qb   = (const float*)d_in[3];
  const float* kvW  = (const float*)d_in[4];
  const float* kvb  = (const float*)d_in[5];
  const float* re   = (const float*)d_in[6];
  const float* ce   = (const float*)d_in[7];
  const float* aggW = (const float*)d_in[8];
  const float* aggb = (const float*)d_in[9];
  const float* n1g  = (const float*)d_in[10];
  const float* n1b  = (const float*)d_in[11];
  const float* l1W  = (const float*)d_in[12];
  const float* l1b  = (const float*)d_in[13];
  const float* l2W  = (const float*)d_in[14];
  const float* l2b  = (const float*)d_in[15];
  const float* n2g  = (const float*)d_in[16];
  const float* n2b  = (const float*)d_in[17];
  const float* outW = (const float*)d_in[18];
  const float* outb = (const float*)d_in[19];

  float* w     = (float*)d_ws;
  float* pe    = w;                       // 1024*256            = 262144
  float* x     = pe + 262144;             // 16*1024*256         = 4194304
  float* kt    = x + 4194304;             // 4*16*1024*256       = 16777216
  float* o     = kt + 16777216;           // 16*1024*1024 (=h1)  = 16777216
  float* tmp   = o + 16777216;            // max(agg/ffn2/recon) = 4456448
  float* pred0 = tmp + 4456448;           // 16384
  float* mm    = pred0 + 16384;           // 2

  float* recon_out = (float*)d_out;                       // 16*272*1024
  float* pred_out  = recon_out + (long long)B_ * DIN_ * N_;

  const int M1 = B_ * N_;        // 16384
  const int M2 = KV_ * B_ * N_;  // 65536

  pe_kernel<<<1024, 256, 0, stream>>>(re, ce, pe);

  // x = query tokens @ qW + qb   (A read strided from NCHW)
  gemm_wmma_bf16<<<dim3(HID_ / 64, M1 / 128), 256, 0, stream>>>(
      qf, N_, (long long)DIN_ * N_, 1LL, (long long)N_,
      qW, HID_, qb, x, HID_, M1, HID_, DIN_, 0);

  // kt = matched tokens @ kvW + kvb
  gemm_wmma_bf16<<<dim3(HID_ / 64, M2 / 128), 256, 0, stream>>>(
      mf, N_, (long long)DIN_ * N_, 1LL, (long long)N_,
      kvW, HID_, kvb, kt, HID_, M2, HID_, DIN_, 0);

  for (int l = 0; l < 2; ++l) {
    attn_kernel<<<dim3(N_, B_), 256, 0, stream>>>(x, kt, pe, o);

    // tmp = o @ aggW[l] + aggb[l]
    gemm_wmma_bf16<<<dim3(HID_ / 64, M1 / 128), 256, 0, stream>>>(
        o, M1, 0LL, (long long)(KV_ * HID_), 1LL,
        aggW + (long long)l * KV_ * HID_ * HID_, HID_, aggb + l * HID_,
        tmp, HID_, M1, HID_, KV_ * HID_, 0);

    add_ln_kernel<<<M1, 256, 0, stream>>>(x, tmp, n1g + l * HID_, n1b + l * HID_);

    // h1 (reuses o) = relu(x @ l1W[l] + l1b[l])
    gemm_wmma_bf16<<<dim3(DFF_ / 64, M1 / 128), 256, 0, stream>>>(
        x, M1, 0LL, (long long)HID_, 1LL,
        l1W + (long long)l * HID_ * DFF_, DFF_, l1b + l * DFF_,
        o, DFF_, M1, DFF_, HID_, 1);

    // tmp = h1 @ l2W[l] + l2b[l]
    gemm_wmma_bf16<<<dim3(HID_ / 64, M1 / 128), 256, 0, stream>>>(
        o, M1, 0LL, (long long)DFF_, 1LL,
        l2W + (long long)l * DFF_ * HID_, HID_, l2b + l * HID_,
        tmp, HID_, M1, HID_, DFF_, 0);

    add_ln_kernel<<<M1, 256, 0, stream>>>(x, tmp, n2g + l * HID_, n2b + l * HID_);
  }

  // recon tokens = x @ outW + outb   (272 -> 5 column blocks with N-tail)
  gemm_wmma_bf16<<<dim3((DIN_ + 63) / 64, M1 / 128), 256, 0, stream>>>(
      x, M1, 0LL, (long long)HID_, 1LL,
      outW, DIN_, outb, tmp, DIN_, M1, DIN_, HID_, 0);

  recon_kernel<<<dim3(N_, B_), 256, 0, stream>>>(tmp, qf, recon_out, pred0);
  minmax_kernel<<<1, 256, 0, stream>>>(pred0, mm);
  resize_kernel<<<(B_ * 256 * 256) / 256, 256, 0, stream>>>(pred0, mm, pred_out);
}
